// MHA_rotary_12369505812817
// MI455X (gfx1250) — matrix operand order repaired and checked by reading, not compiled, not run on hardware
//
#include <hip/hip_runtime.h>
#include <hip/hip_bf16.h>

#define N_HEADS 16
#define HEADSZ  64
#define ROTD    32
#define TSEQ    2048
#define BATCH   2
#define CDIM    1024
#define BTOK    (BATCH*TSEQ)

typedef __attribute__((ext_vector_type(16))) _Float16 v16h;
typedef __attribute__((ext_vector_type(8)))  _Float16 v8h;
typedef __attribute__((ext_vector_type(8)))  float    v8f;
typedef __attribute__((ext_vector_type(4)))  unsigned int u32x4;
typedef __attribute__((ext_vector_type(8)))  unsigned int u32x8;

__device__ __forceinline__ v8f wmma_f32_f16(v16h a, v16h b, v8f c) {
  // D = A(16x32 f16) * B(32x16 f16) + C(16x16 f32)
  return __builtin_amdgcn_wmma_f32_16x16x32_f16(false, a, false, b, (short)0, c,
                                                false, false);
}

// A-fragment (16x32 f16, row-major source, row stride ld halves).
// lane m=lid&15, hi=lid>>4; halves0-7 -> K=hi*8+h ; halves8-15 -> K=16+hi*8+(h-8)
__device__ __forceinline__ v16h load_afrag(const _Float16* p, int ld, int lid) {
  const int m = lid & 15, hi = lid >> 4;
  const _Float16* row = p + (size_t)m * ld;
  v8h lo = *(const v8h*)(row + hi * 8);
  v8h hv = *(const v8h*)(row + 16 + hi * 8);
  v16h r;
#pragma unroll
  for (int i = 0; i < 8; ++i) { r[i] = lo[i]; r[i + 8] = hv[i]; }
  return r;
}

// B-fragment (32x16 f16) from [N][K] row-major source (row stride ld halves).
// lane n=lid&15, hi=lid>>4; halves0-7 -> K=hi*8+h ; halves8-15 -> K=16+hi*8+(h-8), column n.
__device__ __forceinline__ v16h load_bfrag(const _Float16* p, int ld, int lid) {
  const int n = lid & 15, hi = lid >> 4;
  const _Float16* row = p + (size_t)n * ld;
  v8h a = *(const v8h*)(row + hi * 8);
  v8h b = *(const v8h*)(row + 16 + hi * 8);
  v16h r;
#pragma unroll
  for (int i = 0; i < 8; ++i) { r[i] = a[i]; r[i + 8] = b[i]; }
  return r;
}

// ---- Tensor Data Mover: 2D f16 tile (rows x cols) from global -> LDS -------
// D# per CDNA5 ISA §8.3/§8.4. Wave-uniform; issue from one wave only.
__device__ __forceinline__ void tdm_load_2d_f16(unsigned lds_addr,
                                                const void* gptr,
                                                unsigned rows, unsigned cols,
                                                unsigned row_stride_elems) {
  unsigned long long ga = (unsigned long long)gptr;
  u32x4 g0;
  g0[0] = 1u;                                        // count=1 (valid), user mode
  g0[1] = lds_addr;                                  // lds_addr (bytes)
  g0[2] = (unsigned)(ga & 0xffffffffu);              // global_addr[31:0]
  g0[3] = (unsigned)((ga >> 32) & 0x01ffffffu)       // global_addr[56:32]
          | (2u << 30);                              // type = 2 ("image")
  u32x8 g1;
  g1[0] = (1u << 16);                                // wg_mask=0, data_size=1 (2B)
  g1[1] = (cols & 0xffffu) << 16;                    // tensor_dim0[15:0] @ [63:48]
  g1[2] = ((cols >> 16) & 0xffffu)                   // tensor_dim0[31:16]
          | ((rows & 0xffffu) << 16);                // tensor_dim1[15:0]
  g1[3] = ((rows >> 16) & 0xffffu)                   // tensor_dim1[31:16]
          | ((cols & 0xffffu) << 16);                // tile_dim0 @ [127:112]
  g1[4] = (rows & 0xffffu);                          // tile_dim1; tile_dim2=0
  g1[5] = row_stride_elems;                          // tensor_dim0_stride[31:0]
  g1[6] = 0u;                                        // stride[47:32], dim1_stride lo
  g1[7] = 0u;
  asm volatile("tensor_load_to_lds %0, %1" : : "s"(g0), "s"(g1) : "memory");
}

// ---------------- precision conversion kernels ----------------
__global__ __launch_bounds__(256, 1) void k_cvt_x(const float* __restrict__ x,
                                                  _Float16* __restrict__ xh, int n) {
  int i = blockIdx.x * 256 + threadIdx.x;
  if (i < n) xh[i] = (_Float16)x[i];
}

// W is [K=C][N=C] row-major fp32 -> Wt is [N][K] f16, LDS-tiled transpose so both
// global read and write streams are coalesced (avoids ~16x line-granularity blowup).
__global__ __launch_bounds__(256, 1) void k_cvt_wt(const float* __restrict__ W,
                                                   _Float16* __restrict__ Wt) {
  __shared__ _Float16 tile[32][33];
  const int bn = blockIdx.x * 32;   // n base
  const int bk = blockIdx.y * 32;   // k base
  const int tx = threadIdx.x & 31, ty = threadIdx.x >> 5;  // 32x8
#pragma unroll
  for (int i = 0; i < 32; i += 8)
    tile[ty + i][tx] = (_Float16)W[(size_t)(bk + ty + i) * CDIM + bn + tx];
  __syncthreads();
#pragma unroll
  for (int i = 0; i < 32; i += 8)
    Wt[(size_t)(bn + ty + i) * CDIM + bk + tx] = tile[tx][ty + i];
}

// ---------------- fused QKV projection + bias + rotary ----------------
// grid: (BT/128, H, 3), block 128 (4 waves). Wave w: 32 rows x one head (64 cols).
// K-step = 64 per TDM stage -> 16 WMMAs per wave per (wait+2 barriers).
__global__ __launch_bounds__(128, 1) void k_qkv(
    const _Float16* __restrict__ xh,
    const _Float16* __restrict__ wqt, const _Float16* __restrict__ wkt,
    const _Float16* __restrict__ wvt,
    const float* __restrict__ bq, const float* __restrict__ bk,
    const float* __restrict__ bv,
    _Float16* __restrict__ Qh, _Float16* __restrict__ Kh,
    _Float16* __restrict__ Vt) {
  const int which = blockIdx.z;
  const _Float16* wt = (which == 0) ? wqt : (which == 1) ? wkt : wvt;
  const float* bias  = (which == 0) ? bq  : (which == 1) ? bk  : bv;

  const int wave = threadIdx.x >> 5;
  const int lid  = threadIdx.x & 31;
  const int rowBase = blockIdx.x * 128;
  const int head = blockIdx.y;
  const int colBase = head * HEADSZ;

  __shared__ __attribute__((aligned(16))) _Float16 lA[2][128][64];  // 32 KB
  __shared__ __attribute__((aligned(16))) _Float16 lB[2][64][64];   // 16 KB

  v8f acc[2][4] = {};

  const int NK = CDIM / 64;   // 16 stages
  if (wave == 0) {  // prologue: stage 0 -> buf 0
    tdm_load_2d_f16((unsigned)(unsigned long long)&lA[0][0][0],
                    xh + (size_t)rowBase * CDIM, 128, 64, CDIM);
    tdm_load_2d_f16((unsigned)(unsigned long long)&lB[0][0][0],
                    wt + (size_t)colBase * CDIM, 64, 64, CDIM);
  }
  for (int s = 0; s < NK; ++s) {
    const int cur = s & 1, nxt = cur ^ 1;
    if (wave == 0) {
      if (s + 1 < NK) {
        tdm_load_2d_f16((unsigned)(unsigned long long)&lA[nxt][0][0],
                        xh + (size_t)rowBase * CDIM + (s + 1) * 64, 128, 64, CDIM);
        tdm_load_2d_f16((unsigned)(unsigned long long)&lB[nxt][0][0],
                        wt + (size_t)colBase * CDIM + (s + 1) * 64, 64, 64, CDIM);
        __builtin_amdgcn_s_wait_tensorcnt((short)2);  // stage s complete
      } else {
        __builtin_amdgcn_s_wait_tensorcnt((short)0);
      }
    }
    __syncthreads();
#pragma unroll
    for (int kk = 0; kk < 2; ++kk) {
      v16h a0 = load_afrag(&lA[cur][wave * 32][kk * 32], 64, lid);
      v16h a1 = load_afrag(&lA[cur][wave * 32 + 16][kk * 32], 64, lid);
#pragma unroll
      for (int tn = 0; tn < 4; ++tn) {
        v16h b = load_bfrag(&lB[cur][tn * 16][kk * 32], 64, lid);
        acc[0][tn] = wmma_f32_f16(a0, b, acc[0][tn]);
        acc[1][tn] = wmma_f32_f16(a1, b, acc[1][tn]);
      }
    }
    __syncthreads();
  }

  // Epilogue: bias (+ rotary for q/k). Lane holds rows M=r+8*hi, col n per tile.
  const int n = lid & 15, hi = lid >> 4;
  float bb[4];
#pragma unroll
  for (int tn = 0; tn < 4; ++tn) bb[tn] = bias[colBase + tn * 16 + n];
  // inv_freq for frequency index n: 10000^(-2n/32)
  const float invf = __expf(-((float)(2 * n) * (1.0f / ROTD)) * 9.210340371976184f);

#pragma unroll
  for (int mt = 0; mt < 2; ++mt) {
#pragma unroll
    for (int r = 0; r < 8; ++r) {
      int row = rowBase + wave * 32 + mt * 16 + hi * 8 + r;
      int t  = row & (TSEQ - 1);
      int bi = row >> 11;
      float c0 = acc[mt][0][r] + bb[0];
      float c1 = acc[mt][1][r] + bb[1];
      float c2 = acc[mt][2][r] + bb[2];
      float c3 = acc[mt][3][r] + bb[3];
      if (which < 2) {
        // rotary: dims (n, n+16) share angle t*invf
        float sn, cs;
        __sincosf((float)t * invf, &sn, &cs);
        float n0 = c0 * cs - c1 * sn;
        float n1 = c1 * cs + c0 * sn;
        c0 = n0; c1 = n1;
        _Float16* dst = ((which == 0) ? Qh : Kh) +
                        (((size_t)bi * N_HEADS + head) * TSEQ + t) * HEADSZ;
        dst[n]      = (_Float16)c0;
        dst[16 + n] = (_Float16)c1;
        dst[32 + n] = (_Float16)c2;
        dst[48 + n] = (_Float16)c3;
      } else {
        // V stored transposed per head: [hs][T]
        _Float16* dst = Vt + ((size_t)bi * N_HEADS + head) * HEADSZ * TSEQ;
        dst[(size_t)(n)      * TSEQ + t] = (_Float16)c0;
        dst[(size_t)(16 + n) * TSEQ + t] = (_Float16)c1;
        dst[(size_t)(32 + n) * TSEQ + t] = (_Float16)c2;
        dst[(size_t)(48 + n) * TSEQ + t] = (_Float16)c3;
      }
    }
  }
}

// ---------------- causal flash attention ----------------
// grid: (T/64, H, B), block 128. Each wave: one 16-query tile, online softmax
// over 32-key blocks. No block-wide barriers (waves have different trip counts).
// K/V working set (~16MB) is L2-resident (192MB), so direct global frag loads.
__global__ __launch_bounds__(128, 1) void k_attn(
    const _Float16* __restrict__ Qh, const _Float16* __restrict__ Kh,
    const _Float16* __restrict__ Vt, _Float16* __restrict__ AO) {
  const int wave = threadIdx.x >> 5, lid = threadIdx.x & 31;
  const int n = lid & 15, hi = lid >> 4;
  const int head = blockIdx.y, bi = blockIdx.z;
  const int qBase = blockIdx.x * 64 + wave * 16;

  const size_t bh = (size_t)bi * N_HEADS + head;
  const _Float16* Qp = Qh + (bh * TSEQ + qBase) * HEADSZ;
  const _Float16* Kp = Kh + bh * TSEQ * HEADSZ;
  const _Float16* Vp = Vt + bh * HEADSZ * TSEQ;

  __shared__ __attribute__((aligned(16))) _Float16 lP[4][16][32];

  v16h aq0 = load_afrag(Qp, HEADSZ, lid);        // d = 0..31
  v16h aq1 = load_afrag(Qp + 32, HEADSZ, lid);   // d = 32..63

  v8f o[4] = {};
  float mrow[8], lrow[8];
#pragma unroll
  for (int r = 0; r < 8; ++r) { mrow[r] = -1e30f; lrow[r] = 0.0f; }

  const int nIter = (qBase + 16 + 31) >> 5;  // 32-key blocks needed (causal)
  for (int it = 0; it < nIter; ++it) {
    const int kB = it * 32;
    if (it + 1 < nIter) {
      __builtin_prefetch(Kp + (size_t)(kB + 32) * HEADSZ + (size_t)lid * 64, 0, 1);
      __builtin_prefetch(Vp + (size_t)(lid * 2) * TSEQ + kB + 32, 0, 1);
    }
    v8f s0 = {}, s1 = {};
    {
      v16h b00 = load_bfrag(Kp + (size_t)(kB)*HEADSZ,          HEADSZ, lid);
      v16h b01 = load_bfrag(Kp + (size_t)(kB)*HEADSZ + 32,     HEADSZ, lid);
      s0 = wmma_f32_f16(aq0, b00, s0);
      s0 = wmma_f32_f16(aq1, b01, s0);
      v16h b10 = load_bfrag(Kp + (size_t)(kB + 16) * HEADSZ,      HEADSZ, lid);
      v16h b11 = load_bfrag(Kp + (size_t)(kB + 16) * HEADSZ + 32, HEADSZ, lid);
      s1 = wmma_f32_f16(aq0, b10, s1);
      s1 = wmma_f32_f16(aq1, b11, s1);
    }
    const bool edge = (kB + 32 > qBase);
#pragma unroll
    for (int r = 0; r < 8; ++r) {
      float v0 = s0[r] * 0.125f, v1 = s1[r] * 0.125f;  // 1/sqrt(64)
      if (edge) {
        int q = qBase + hi * 8 + r;
        if (kB + n      > q) v0 = -1e30f;
        if (kB + 16 + n > q) v1 = -1e30f;
      }
      // row-wise reductions: row M lives in one 16-lane half -> xor masks 1..8
      float smax = fmaxf(v0, v1);
#pragma unroll
      for (int msk = 1; msk < 16; msk <<= 1) smax = fmaxf(smax, __shfl_xor(smax, msk, 32));
      float mnew = fmaxf(mrow[r], smax);
      float corr = __expf(mrow[r] - mnew);
      float p0 = __expf(v0 - mnew), p1 = __expf(v1 - mnew);
      float ps = p0 + p1;
#pragma unroll
      for (int msk = 1; msk < 16; msk <<= 1) ps += __shfl_xor(ps, msk, 32);
      lrow[r] = lrow[r] * corr + ps;
      mrow[r] = mnew;
#pragma unroll
      for (int tn = 0; tn < 4; ++tn) o[tn][r] *= corr;
      int M = hi * 8 + r;
      lP[wave][M][n]      = (_Float16)p0;
      lP[wave][M][16 + n] = (_Float16)p1;
    }
    // same-wave DS write -> cross-lane DS read: DS ops are in-order per wave,
    // but force the wait + stop compiler reordering.
    asm volatile("s_wait_dscnt 0" ::: "memory");
    v16h pa = load_afrag(&lP[wave][0][0], 32, lid);
#pragma unroll
    for (int tn = 0; tn < 4; ++tn) {
      v16h bv = load_bfrag(Vp + (size_t)(tn * 16) * TSEQ + kB, TSEQ, lid);
      o[tn] = wmma_f32_f16(pa, bv, o[tn]);
    }
  }

#pragma unroll
  for (int r = 0; r < 8; ++r) {
    int q = qBase + hi * 8 + r;
    float invl = 1.0f / lrow[r];
    _Float16* dst = AO + ((size_t)bi * TSEQ + q) * CDIM + head * HEADSZ;
#pragma unroll
    for (int tn = 0; tn < 4; ++tn)
      dst[tn * 16 + n] = (_Float16)(o[tn][r] * invl);
  }
}

// ---------------- output projection (fp32 out + bias) ----------------
// grid: (BT/128, C/64), block 128. Same 32x64-per-wave tiling + TDM staging.
__global__ __launch_bounds__(128, 1) void k_oproj(
    const _Float16* __restrict__ AO, const _Float16* __restrict__ wot,
    const float* __restrict__ bo, float* __restrict__ out) {
  const int wave = threadIdx.x >> 5, lid = threadIdx.x & 31;
  const int rowBase = blockIdx.x * 128;
  const int colBase = blockIdx.y * 64;
  __shared__ __attribute__((aligned(16))) _Float16 lA[2][128][64];
  __shared__ __attribute__((aligned(16))) _Float16 lB[2][64][64];
  v8f acc[2][4] = {};

  const int NK = CDIM / 64;
  if (wave == 0) {
    tdm_load_2d_f16((unsigned)(unsigned long long)&lA[0][0][0],
                    AO + (size_t)rowBase * CDIM, 128, 64, CDIM);
    tdm_load_2d_f16((unsigned)(unsigned long long)&lB[0][0][0],
                    wot + (size_t)colBase * CDIM, 64, 64, CDIM);
  }
  for (int s = 0; s < NK; ++s) {
    const int cur = s & 1, nxt = cur ^ 1;
    if (wave == 0) {
      if (s + 1 < NK) {
        tdm_load_2d_f16((unsigned)(unsigned long long)&lA[nxt][0][0],
                        AO + (size_t)rowBase * CDIM + (s + 1) * 64, 128, 64, CDIM);
        tdm_load_2d_f16((unsigned)(unsigned long long)&lB[nxt][0][0],
                        wot + (size_t)colBase * CDIM + (s + 1) * 64, 64, 64, CDIM);
        __builtin_amdgcn_s_wait_tensorcnt((short)2);
      } else {
        __builtin_amdgcn_s_wait_tensorcnt((short)0);
      }
    }
    __syncthreads();
#pragma unroll
    for (int kk = 0; kk < 2; ++kk) {
      v16h a0 = load_afrag(&lA[cur][wave * 32][kk * 32], 64, lid);
      v16h a1 = load_afrag(&lA[cur][wave * 32 + 16][kk * 32], 64, lid);
#pragma unroll
      for (int tn = 0; tn < 4; ++tn) {
        v16h b = load_bfrag(&lB[cur][tn * 16][kk * 32], 64, lid);
        acc[0][tn] = wmma_f32_f16(a0, b, acc[0][tn]);
        acc[1][tn] = wmma_f32_f16(a1, b, acc[1][tn]);
      }
    }
    __syncthreads();
  }
  const int n = lid & 15, hi = lid >> 4;
#pragma unroll
  for (int tn = 0; tn < 4; ++tn) {
    float bb = bo[colBase + tn * 16 + n];
#pragma unroll
    for (int mt = 0; mt < 2; ++mt) {
#pragma unroll
      for (int r = 0; r < 8; ++r) {
        int row = rowBase + wave * 32 + mt * 16 + hi * 8 + r;
        out[(size_t)row * CDIM + colBase + tn * 16 + n] = acc[mt][tn][r] + bb;
      }
    }
  }
}

extern "C" void kernel_launch(void* const* d_in, const int* in_sizes, int n_in,
                              void* d_out, int out_size, void* d_ws, size_t ws_size,
                              hipStream_t stream) {
  (void)in_sizes; (void)n_in; (void)out_size; (void)ws_size;
  const float* x  = (const float*)d_in[0];
  const float* Wq = (const float*)d_in[1];
  const float* bq = (const float*)d_in[2];
  const float* Wk = (const float*)d_in[3];
  const float* bk = (const float*)d_in[4];
  const float* Wv = (const float*)d_in[5];
  const float* bv = (const float*)d_in[6];
  const float* Wo = (const float*)d_in[7];
  const float* bo = (const float*)d_in[8];
  float* out = (float*)d_out;

  char* w = (char*)d_ws;
  size_t off = 0;
  auto alloc = [&](size_t bytes) {
    char* p = w + off;
    off += (bytes + 255) & ~(size_t)255;
    return p;
  };
  _Float16* xh  = (_Float16*)alloc((size_t)BTOK * CDIM * 2);
  _Float16* wqt = (_Float16*)alloc((size_t)CDIM * CDIM * 2);
  _Float16* wkt = (_Float16*)alloc((size_t)CDIM * CDIM * 2);
  _Float16* wvt = (_Float16*)alloc((size_t)CDIM * CDIM * 2);
  _Float16* wot = (_Float16*)alloc((size_t)CDIM * CDIM * 2);
  _Float16* Qh  = (_Float16*)alloc((size_t)BTOK * CDIM * 2);
  _Float16* Kh  = (_Float16*)alloc((size_t)BTOK * CDIM * 2);
  _Float16* Vt  = (_Float16*)alloc((size_t)BTOK * CDIM * 2);
  _Float16* AO  = (_Float16*)alloc((size_t)BTOK * CDIM * 2);

  k_cvt_x<<<(BTOK * CDIM) / 256, 256, 0, stream>>>(x, xh, BTOK * CDIM);
  dim3 tg(CDIM / 32, CDIM / 32);
  k_cvt_wt<<<tg, 256, 0, stream>>>(Wq, wqt);
  k_cvt_wt<<<tg, 256, 0, stream>>>(Wk, wkt);
  k_cvt_wt<<<tg, 256, 0, stream>>>(Wv, wvt);
  k_cvt_wt<<<tg, 256, 0, stream>>>(Wo, wot);

  k_qkv<<<dim3(BTOK / 128, N_HEADS, 3), 128, 0, stream>>>(xh, wqt, wkt, wvt,
                                                          bq, bk, bv, Qh, Kh, Vt);
  k_attn<<<dim3(TSEQ / 64, N_HEADS, BATCH), 128, 0, stream>>>(Qh, Kh, Vt, AO);
  k_oproj<<<dim3(BTOK / 128, CDIM / 64), 128, 0, stream>>>(AO, wot, bo, out);
}
